// ConvLSTM_18545668784167
// MI455X (gfx1250) — matrix operand order repaired and checked by reading, not compiled, not run on hardware
//
#include <hip/hip_runtime.h>
#include <hip/hip_bf16.h>

// ConvLSTM on MI455X (gfx1250): implicit-GEMM conv via bf16 WMMA + fused gates.
//
// Shapes: X(16,8,20,64,64) f32, W(128,40,3,3) f32, b(128), W_ci/cf/co(32,64,64)
// Out: (16,32,20,64,64) f32.
//
// Per timestep GEMM: M = pixels, N = 128 gates, K = 40ch*9taps = 360.
// Custom K-permutation pads to 12 chunks of 32 for v_wmma_f32_16x16x32_bf16
// while making A-fragment LDS offsets compile-time immediates:
//   physical p -> u = (p&7) + ((p&16)>>1), g = (p>>3)&1
//   logical    -> s = kc*16 + u; s<180: c = g*20 + s/9, tap = s%9; else pad(0)
// A lane layout (p = (i&8)*2 + half*8 + (i&7)) gives u==i, g==half, so
//   lds_off = (apy*10 + apx) + half*2000 + OFF(kc*16 + i), OFF constant.
// Pad slots read LDS offset 0; the matching B entries are packed as zero.
// B fragments are software double-buffered across the kc loop so each WMMA
// consumes loads issued one iteration earlier (A-gather hides the latency).

typedef __attribute__((ext_vector_type(16))) __bf16 v16bf;
typedef __attribute__((ext_vector_type(8)))  float  v8f;

#define B_    16
#define CIN_  8
#define T_    20
#define HW_   64
#define COUT_ 32
#define SREAL 180   // real logical slots per channel-group half (20ch * 9taps)

__device__ __forceinline__ float sigf(float x) {
    return 1.0f / (1.0f + __expf(-x));
}
__device__ __forceinline__ float tanh_fast(float x) {
    float e = __expf(-2.0f * fabsf(x));
    float r = (1.0f - e) / (1.0f + e);
    return copysignf(r, x);
}

// Constant-foldable im2col LDS offset for logical slot s in [0,180):
//   c_local = s/9, tap = s%9  ->  c_local*100 + dy*10 + dx
__device__ __forceinline__ int a_off(int s) {
    int cl  = s / 9;
    int tap = s - cl * 9;
    int dy  = tap / 3;
    int dx  = tap - dy * 3;
    return cl * 100 + dy * 10 + dx;
}

// ---------------------------------------------------------------------------
// Pack W (128, 40, 3, 3) f32 -> Wpk bf16 in WMMA B-fragment order under the
// custom K-permutation. Entry (kc, n, half, i):
//   physical p = (i&8)*2 + half*8 + (i&7)  ->  u = i, g = half
//   s = kc*16 + u;  s<180: weight W[n][g*20 + s/9][s%9], else 0.
// ---------------------------------------------------------------------------
__global__ __launch_bounds__(256) void ConvLSTM_prep_weights(
    const float* __restrict__ W, __bf16* __restrict__ Wpk)
{
    int idx = blockIdx.x * blockDim.x + threadIdx.x;
    if (idx >= 12 * 128 * 32) return;
    int i    = idx & 15;
    int half = (idx >> 4) & 1;
    int n    = (idx >> 5) & 127;
    int kc   = idx >> 12;

    int u = i;
    int g = half;
    int s = kc * 16 + u;

    float v = 0.0f;
    if (s < SREAL) {
        int cl  = s / 9;
        int tap = s - cl * 9;
        int c   = g * 20 + cl;
        v = W[(n * 40 + c) * 9 + tap];
    }
    Wpk[idx] = (__bf16)v;
}

__global__ __launch_bounds__(256) void ConvLSTM_zero(float* __restrict__ p, int n)
{
    int idx = blockIdx.x * blockDim.x + threadIdx.x;
    if (idx < n) p[idx] = 0.0f;
}

// ---------------------------------------------------------------------------
// One timestep. Grid: 1024 blocks (16 batch * 8x8 tiles of 8x8 pixels),
// 256 threads = 8 wave32. Wave (wm, wn): wm picks 16-pixel M-subtile,
// wn picks 64-channel N-half; 4 accumulators (16x16 f32) per wave.
// ---------------------------------------------------------------------------
__global__ __launch_bounds__(256) void ConvLSTM_step(
    const float*  __restrict__ X,
    const __bf16* __restrict__ Wpk,
    const float*  __restrict__ bias,
    const float*  __restrict__ Wci,
    const float*  __restrict__ Wcf,
    const float*  __restrict__ Wco,
    float* __restrict__ hbuf,   // (B, 32, 64, 64)
    float* __restrict__ cbuf,   // (B, 32, 64, 64)
    float* __restrict__ out,    // (B, 32, 20, 64, 64)
    int t)
{
    __shared__ __bf16 lds_in[40 * 100];   // [c][10][10] halo tile, 8 KB
    __shared__ float  gbuf[64 * 128];     // 64 px x 128 gates, 32 KB

    const int tid  = threadIdx.x;
    const int blk  = blockIdx.x;
    const int bb   = blk >> 6;
    const int tile = blk & 63;
    const int y0   = (tile >> 3) << 3;
    const int x0   = (tile & 7) << 3;

    // ---- Stage input halo (x_t channels 0..7, h channels 8..39) as bf16 ----
    for (int idx = tid; idx < 40 * 100; idx += 256) {
        int c   = idx / 100;
        int rem = idx - c * 100;
        int ry  = rem / 10;
        int rx  = rem - ry * 10;
        int gy  = y0 + ry - 1;
        int gx  = x0 + rx - 1;
        float v = 0.0f;
        if (gy >= 0 && gy < HW_ && gx >= 0 && gx < HW_) {
            if (c < CIN_)
                v = X[(((bb * CIN_ + c) * T_ + t) * HW_ + gy) * HW_ + gx];
            else
                v = hbuf[((bb * COUT_ + (c - CIN_)) * HW_ + gy) * HW_ + gx];
        }
        lds_in[idx] = (__bf16)v;
    }
    __syncthreads();

    const int wave = tid >> 5;
    const int lane = tid & 31;
    const int half = lane >> 4;      // K-half selector in A/B lane layouts
    const int col  = lane & 15;      // M row (A) / N column (B, C/D)
    const int wm   = wave >> 1;      // M-subtile 0..3
    const int wn   = wave & 1;       // N-half 0..1

    // Pixel row for A-matrix row m=col of this wave's 16-row subtile.
    const int apy = wm * 2 + (col >> 3);
    const int apx = col & 7;
    // Single lane-dependent base; all per-element offsets are immediates.
    const int abase = apy * 10 + apx + half * 2000;

    // Per-lane base into packed weights for this wave's N-half.
    const __bf16* wbase = Wpk + (((wn * 64 + col) * 2 + half) << 4);

    v8f acc[4];
    #pragma unroll
    for (int nt = 0; nt < 4; nt++) acc[nt] = (v8f){0,0,0,0,0,0,0,0};

    // ---- Prime B double-buffer with chunk 0 ----
    v16bf bcur[4], bnxt[4];
    #pragma unroll
    for (int nt = 0; nt < 4; nt++)
        bcur[nt] = *(const v16bf*)(wbase + ((nt * 16 * 2) << 4));

    #pragma unroll
    for (int kc = 0; kc < 12; kc++) {
        // ---- Prefetch next chunk's B fragments (32 B contiguous per lane) --
        if (kc < 11) {
            #pragma unroll
            for (int nt = 0; nt < 4; nt++)
                bnxt[nt] = *(const v16bf*)(
                    wbase + ((((kc + 1) * 128 + nt * 16) * 2) << 4));
        }
        // ---- A fragment: 16 LDS loads at constant offsets from abase ----
        v16bf a;
        #pragma unroll
        for (int i = 0; i < 16; i++) {
            int s   = kc * 16 + i;                 // compile-time after unroll
            int off = (s < SREAL) ? a_off(s) : 0;  // pad -> B is zero anyway
            a[i] = lds_in[abase + off];
        }
        // ---- 4 WMMAs against the buffered B fragments ----
        #pragma unroll
        for (int nt = 0; nt < 4; nt++) {
            acc[nt] = __builtin_amdgcn_wmma_f32_16x16x32_bf16(
                false, a, false, bcur[nt], (short)0, acc[nt], false, false);
        }
        #pragma unroll
        for (int nt = 0; nt < 4; nt++) bcur[nt] = bnxt[nt];
    }

    // ---- Spill accumulators to LDS for cross-wave gate exchange ----
    // C/D layout: VGPR r -> M = r + 8*half, N = col.
    {
        int gb = (wm * 16 + half * 8) * 128 + wn * 64 + col;
        #pragma unroll
        for (int nt = 0; nt < 4; nt++) {
            #pragma unroll
            for (int r = 0; r < 8; r++) {
                gbuf[gb + r * 128 + nt * 16] = acc[nt][r];
            }
        }
    }
    __syncthreads();

    // ---- Fused LSTM gate math: 64 px x 32 cout = 2048 elements ----
    for (int e = tid; e < 64 * COUT_; e += 256) {
        int m  = e & 63;
        int co = e >> 6;
        int y  = y0 + (m >> 3);
        int x  = x0 + (m & 7);

        float gi = gbuf[m * 128 + co     ] + bias[co     ];
        float gf = gbuf[m * 128 + co + 32] + bias[co + 32];
        float gc = gbuf[m * 128 + co + 64] + bias[co + 64];
        float go = gbuf[m * 128 + co + 96] + bias[co + 96];

        int ci = ((bb * COUT_ + co) * HW_ + y) * HW_ + x;
        int wi = (co * HW_ + y) * HW_ + x;

        float cp  = cbuf[ci];
        float i_a = sigf(gi + Wci[wi] * cp);
        float f_a = sigf(gf + Wcf[wi] * cp);
        float cn  = f_a * cp + i_a * tanh_fast(gc);
        float o_a = sigf(go + Wco[wi] * cn);
        float hn  = o_a * tanh_fast(cn);

        cbuf[ci] = cn;
        hbuf[ci] = hn;
        out[(((bb * COUT_ + co) * T_ + t) * HW_ + y) * HW_ + x] = hn;
    }
}

// ---------------------------------------------------------------------------
extern "C" void kernel_launch(void* const* d_in, const int* in_sizes, int n_in,
                              void* d_out, int out_size, void* d_ws, size_t ws_size,
                              hipStream_t stream)
{
    (void)in_sizes; (void)n_in; (void)out_size; (void)ws_size;

    const float* X   = (const float*)d_in[0];
    const float* W   = (const float*)d_in[1];
    const float* b   = (const float*)d_in[2];
    const float* Wci = (const float*)d_in[3];
    const float* Wcf = (const float*)d_in[4];
    const float* Wco = (const float*)d_in[5];
    float* out = (float*)d_out;

    // Workspace layout:
    //   [0, 98304)               : Wpk, 12*128*32 bf16
    //   [98304, +8388608)        : hbuf (B,32,64,64) f32
    //   [next,  +8388608)        : cbuf (B,32,64,64) f32
    char* ws = (char*)d_ws;
    __bf16* Wpk  = (__bf16*)ws;
    float*  hbuf = (float*)(ws + 98304);
    float*  cbuf = (float*)(ws + 98304 + 8388608);

    ConvLSTM_prep_weights<<<(12 * 128 * 32 + 255) / 256, 256, 0, stream>>>(W, Wpk);

    int state_elems = 2 * B_ * COUT_ * HW_ * HW_;
    ConvLSTM_zero<<<(state_elems + 255) / 256, 256, 0, stream>>>(hbuf, state_elems);

    for (int t = 0; t < T_; t++) {
        ConvLSTM_step<<<B_ * 64, 256, 0, stream>>>(
            X, Wpk, b, Wci, Wcf, Wco, hbuf, cbuf, out, t);
    }
}
